// ContextualBanditRouter_18339510354409
// MI455X (gfx1250) — matrix operand order repaired and checked by reading, not compiled, not run on hardware
//
#include <hip/hip_runtime.h>
#include <hip/hip_bf16.h>

// ---------------------------------------------------------------------------
// ContextualBanditRouter for MI455X (gfx1250, wave32, WMMA bf16 16x16x32)
// ---------------------------------------------------------------------------
#define BB 16384
#define DD 256
#define EE 8
#define OO 128
#define CCTX 32

#define RW_OFF   (BB * OO)              // routing_w  [B,2]
#define TI_OFF   (RW_OFF + 2 * BB)      // top_idx    [B,2] (int32 bits)
#define ST_OFF   (TI_OFF + 2 * BB)      // new_pulls[8], new_avg[8], new_ema[8]

typedef __bf16 bf16_t;
typedef bf16_t v16bf __attribute__((ext_vector_type(16)));
typedef float  v8f   __attribute__((ext_vector_type(8)));

static __device__ inline v8f wmma_bf16(v16bf a, v16bf b, v8f c) {
  // D = A(16x32 bf16) * B(32x16 bf16) + C(16x16 f32)
  return __builtin_amdgcn_wmma_f32_16x16x32_bf16(
      /*neg_a=*/false, a, /*neg_b=*/false, b,
      /*c_mod=*/(short)0, c, /*reuse_a=*/false, /*reuse_b=*/false);
}

// A fragment (16x32, MxK) from row-major f32 global memory, cvt->bf16.
// Lane mapping: m = lane&15; K runs: [half*8 .. half*8+7] and [+16 .. +23].
static __device__ inline v16bf a_frag_global_f32(const float* base, int row0,
                                                 int ld, int kbase, int lane) {
  int m = lane & 15, half = lane >> 4;
  const float* p = base + (row0 + m) * ld + kbase + half * 8;
  v16bf a;
#pragma unroll
  for (int j = 0; j < 8; ++j) {
    a[j]     = (bf16_t)p[j];
    a[j + 8] = (bf16_t)p[16 + j];
  }
  return a;
}

// A fragment from row-major bf16 memory (LDS or global).
static __device__ inline v16bf a_frag_bf16(const bf16_t* base, int ld,
                                           int kbase, int lane) {
  int m = lane & 15, half = lane >> 4;
  const bf16_t* p = base + m * ld + kbase + half * 8;
  v16bf a;
#pragma unroll
  for (int j = 0; j < 8; ++j) {
    a[j]     = p[j];
    a[j + 8] = p[16 + j];
  }
  return a;
}

// B fragment (32x16, KxN) from an N-major (transposed) bf16 matrix BT[n][k].
// Lane mapping: n = lane&15 (+16*ntile); K runs: [half*8 .. half*8+7] and [+16 .. +23].
static __device__ inline v16bf b_frag_bf16(const bf16_t* bt, int ld, int ntile,
                                           int kbase, int lane) {
  int n = (lane & 15) + ntile * 16, half = lane >> 4;
  const bf16_t* p = bt + n * ld + kbase + half * 8;
  v16bf b;
#pragma unroll
  for (int j = 0; j < 8; ++j) {
    b[j]     = p[j];
    b[j + 8] = p[16 + j];
  }
  return b;
}

// ---------------------------------------------------------------------------
// Kernel 0: repack We [E][D][O] f32  ->  WeT [E][O][D] bf16 (in d_ws)
// ---------------------------------------------------------------------------
__global__ __launch_bounds__(256) void k_transpose_we(const float* __restrict__ We,
                                                      bf16_t* __restrict__ WeT) {
  int i = blockIdx.x * 256 + threadIdx.x;  // i < E*O*D
  int d = i & (DD - 1);
  int o = (i >> 8) & (OO - 1);
  int e = i >> 15;
  WeT[i] = (bf16_t)We[(e * DD + d) * OO + o];
}

// ---------------------------------------------------------------------------
// Kernel 1: router MLP + UCB + top-2 + softmax.  4 waves -> 64 rows / block.
// ---------------------------------------------------------------------------
__global__ __launch_bounds__(128) void k_router(
    const float* __restrict__ x, const float* __restrict__ W1,
    const float* __restrict__ b1, const float* __restrict__ W2,
    const float* __restrict__ b2, const float* __restrict__ S1,
    const float* __restrict__ sb1, const float* __restrict__ S2,
    const float* __restrict__ sb2, const float* __restrict__ rew,
    const int* __restrict__ pulls, const int* __restrict__ total_sel,
    float* __restrict__ out, float* __restrict__ ws) {
  __shared__ bf16_t W1T[64 * 256];   // W1^T  [n][k]
  __shared__ bf16_t W2T[32 * 64];    // W2^T
  __shared__ bf16_t S1T[32 * 32];    // S1^T
  __shared__ bf16_t S2T[16 * 32];    // S2^T, cols 8..15 zero-padded
  __shared__ float b1s[64], b2s[32], sb1s[32], sb2s[8];
  __shared__ float ucb_add[8];
  __shared__ float acc16[16];        // [0..7] counts, [8..15] prob sums
  __shared__ bf16_t hstage[4][16 * 64];
  __shared__ bf16_t cstage[4][16 * 32];
  __shared__ bf16_t rstage[4][16 * 32];
  __shared__ float  sstage[4][16 * 8];

  int tid = threadIdx.x, lane = tid & 31, wid = tid >> 5;
  int rowbase = blockIdx.x * 64 + wid * 16;

  // Warm L2/L0 for this wave's x rows while we stage the weights (CDNA5
  // global_prefetch_b8). One row = 1 KB = 16 cachelines; touch 4 segments.
  {
    const float* xr = x + (rowbase + (lane & 15)) * DD;
#pragma unroll
    for (int s = 0; s < 4; ++s) __builtin_prefetch(xr + s * 64, 0, 3);
  }

  for (int i = tid; i < 64 * 256; i += 128) {
    int n = i >> 8, d = i & 255;
    W1T[i] = (bf16_t)W1[d * 64 + n];
  }
  for (int i = tid; i < 32 * 64; i += 128) {
    int n = i >> 6, d = i & 63;
    W2T[i] = (bf16_t)W2[d * 32 + n];
  }
  for (int i = tid; i < 32 * 32; i += 128) {
    int n = i >> 5, k = i & 31;
    S1T[i] = (bf16_t)S1[k * 32 + n];
  }
  for (int i = tid; i < 16 * 32; i += 128) {
    int n = i >> 5, k = i & 31;
    S2T[i] = (n < 8) ? (bf16_t)S2[k * 8 + n] : (bf16_t)0.0f;
  }
  if (tid < 64) b1s[tid] = b1[tid];
  if (tid < 32) b2s[tid] = b2[tid];
  if (tid < 32) sb1s[tid] = sb1[tid];
  if (tid < 8)  sb2s[tid] = sb2[tid];
  if (tid < 16) acc16[tid] = 0.0f;
  if (tid < 8) {
    float pf = (float)pulls[tid];
    float sp = fmaxf(pf, 1.0f);
    float explo = 0.1f * sqrtf(logf((float)total_sel[0]) / sp);
    float avg = (pf > 0.0f) ? rew[tid] / sp : 0.0f;
    ucb_add[tid] = avg + explo;
  }
  __syncthreads();

  int half = lane >> 4;

  // ---- stage 1: h = relu(x @ W1 + b1)  [16x64] ----
  v16bf a[8];
#pragma unroll
  for (int ks = 0; ks < 8; ++ks)
    a[ks] = a_frag_global_f32(x, rowbase, DD, ks * 32, lane);
  bf16_t* hst = hstage[wid];
#pragma unroll
  for (int nt = 0; nt < 4; ++nt) {
    v8f acc = {};
#pragma unroll
    for (int ks = 0; ks < 8; ++ks)
      acc = wmma_bf16(a[ks], b_frag_bf16(W1T, 256, nt, ks * 32, lane), acc);
    int col = (lane & 15) + nt * 16;
#pragma unroll
    for (int r = 0; r < 8; ++r) {
      int row = r + 8 * half;
      hst[row * 64 + col] = (bf16_t)fmaxf(acc[r] + b1s[col], 0.0f);
    }
  }

  // ---- stage 2: context = tanh(h @ W2 + b2)  [16x32] ----
  v16bf a20 = a_frag_bf16(hst, 64, 0, lane);
  v16bf a21 = a_frag_bf16(hst, 64, 32, lane);
  bf16_t* cst = cstage[wid];
#pragma unroll
  for (int nt = 0; nt < 2; ++nt) {
    v8f acc = {};
    acc = wmma_bf16(a20, b_frag_bf16(W2T, 64, nt, 0, lane), acc);
    acc = wmma_bf16(a21, b_frag_bf16(W2T, 64, nt, 32, lane), acc);
    int col = (lane & 15) + nt * 16;
#pragma unroll
    for (int r = 0; r < 8; ++r) {
      int row = r + 8 * half;
      cst[row * 32 + col] = (bf16_t)tanhf(acc[r] + b2s[col]);
    }
  }

  // ---- stage 3: r1 = relu(context @ S1 + sb1)  [16x32] ----
  v16bf a3 = a_frag_bf16(cst, 32, 0, lane);
  bf16_t* rst = rstage[wid];
#pragma unroll
  for (int nt = 0; nt < 2; ++nt) {
    v8f acc = {};
    acc = wmma_bf16(a3, b_frag_bf16(S1T, 32, nt, 0, lane), acc);
    int col = (lane & 15) + nt * 16;
#pragma unroll
    for (int r = 0; r < 8; ++r) {
      int row = r + 8 * half;
      rst[row * 32 + col] = (bf16_t)fmaxf(acc[r] + sb1s[col], 0.0f);
    }
  }

  // ---- stage 4: scores = r1 @ S2 + sb2  [16x8] (N padded to 16) ----
  {
    v16bf a4 = a_frag_bf16(rst, 32, 0, lane);
    v8f acc = {};
    acc = wmma_bf16(a4, b_frag_bf16(S2T, 32, 0, 0, lane), acc);
    int col = lane & 15;
    if (col < 8) {
#pragma unroll
      for (int r = 0; r < 8; ++r) {
        int row = r + 8 * half;
        sstage[wid][row * 8 + col] = acc[r] + sb2s[col];
      }
    }
  }

  // ---- stage 5: UCB + top-2 + softmax (lanes 0..15 <-> 16 rows) ----
  if (lane < 16) {
    int row = lane;
    float u[8];
#pragma unroll
    for (int e = 0; e < 8; ++e) u[e] = sstage[wid][row * 8 + e] + ucb_add[e];
    int i0 = 0; float v0 = u[0];
#pragma unroll
    for (int e = 1; e < 8; ++e)
      if (u[e] > v0) { v0 = u[e]; i0 = e; }
    int i1 = (i0 == 0) ? 1 : 0; float v1 = u[i1];
#pragma unroll
    for (int e = 0; e < 8; ++e)
      if (e != i0 && u[e] > v1) { v1 = u[e]; i1 = e; }
    float e1 = expf(v1 - v0);          // v0 >= v1, stable softmax
    float inv = 1.0f / (1.0f + e1);
    float w0 = inv, w1 = e1 * inv;
    int grow = rowbase + row;
    out[RW_OFF + grow * 2 + 0] = w0;
    out[RW_OFF + grow * 2 + 1] = w1;
    ((int*)out)[TI_OFF + grow * 2 + 0] = i0;
    ((int*)out)[TI_OFF + grow * 2 + 1] = i1;
    atomicAdd(&acc16[i0], 1.0f);
    atomicAdd(&acc16[i1], 1.0f);
    atomicAdd(&acc16[8 + i0], w0);
    atomicAdd(&acc16[8 + i1], w1);
  }
  __syncthreads();
  if (tid < 8) {
    atomicAdd(&ws[tid], acc16[tid]);            // counts
    atomicAdd(&ws[16 + tid], acc16[8 + tid]);   // routing-prob sums
  }
}

// ---------------------------------------------------------------------------
// Kernel 2: expert GEMMs + predictions + rewards. 8 waves <-> 8 experts,
// 32 rows per block.
// ---------------------------------------------------------------------------
__global__ __launch_bounds__(256) void k_expert(
    const float* __restrict__ x, const float* __restrict__ targets,
    const bf16_t* __restrict__ WeT, const float* __restrict__ be,
    float* __restrict__ out, float* __restrict__ ws) {
  __shared__ bf16_t xs[32 * 256];    // x tile, bf16
  __shared__ float tg[32 * 128];     // targets tile
  __shared__ float bes[8 * 128];     // expert biases
  __shared__ float predl[32 * 128];  // prediction accumulator
  __shared__ float selw[64];         // routing weights  [32][2]
  __shared__ int   seli[64];         // selected experts [32][2]
  __shared__ float rsuml[8];         // per-expert reward sums

  int tid = threadIdx.x, lane = tid & 31, w = tid >> 5;
  int base = blockIdx.x * 32;
  int half = lane >> 4, n = lane & 15;

  const bf16_t* wt = WeT + w * (OO * DD);  // this wave's expert, [o][d] bf16

  // Warm L2 for this lane's 8 weight-panel rows (CDNA5 global_prefetch_b8)
  // while the block stages x/targets into LDS below.
#pragma unroll
  for (int nt = 0; nt < 8; ++nt)
    __builtin_prefetch(wt + (nt * 16 + n) * DD + half * 128, 0, 3);

  for (int i = tid; i < 32 * 256; i += 256) {
    int r = i >> 8, d = i & 255;
    xs[i] = (bf16_t)x[(base + r) * DD + d];
  }
  for (int i = tid; i < 32 * 128; i += 256) {
    int r = i >> 7, o = i & 127;
    tg[i] = targets[(base + r) * OO + o];
    predl[i] = 0.0f;
  }
  for (int i = tid; i < 8 * 128; i += 256) bes[i] = be[i];
  if (tid < 64) {
    int r = tid >> 1, k = tid & 1;
    selw[tid] = out[RW_OFF + (base + r) * 2 + k];
    seli[tid] = ((const int*)out)[TI_OFF + (base + r) * 2 + k];
  }
  if (tid < 8) rsuml[tid] = 0.0f;
  __syncthreads();

#pragma unroll
  for (int t = 0; t < 2; ++t) {  // two 16-row tiles
    const bf16_t* xrow = xs + t * 16 * 256;
    v16bf a[8];
#pragma unroll
    for (int ks = 0; ks < 8; ++ks) a[ks] = a_frag_bf16(xrow, 256, ks * 32, lane);

    float sq[8] = {0.f, 0.f, 0.f, 0.f, 0.f, 0.f, 0.f, 0.f};
#pragma unroll
    for (int nt = 0; nt < 8; ++nt) {
      v8f acc = {};
#pragma unroll
      for (int ks = 0; ks < 8; ++ks)
        acc = wmma_bf16(a[ks], b_frag_bf16(wt, 256, nt, ks * 32, lane), acc);
      int o = nt * 16 + n;
#pragma unroll
      for (int r = 0; r < 8; ++r) {
        int row = t * 16 + r + 8 * half;
        float val = acc[r] + bes[w * 128 + o];
        float dd = val - tg[row * 128 + o];
        sq[r] += dd * dd;
        float wsum = 0.0f;
        if (seli[row * 2 + 0] == w) wsum += selw[row * 2 + 0];
        if (seli[row * 2 + 1] == w) wsum += selw[row * 2 + 1];
        if (wsum != 0.0f) atomicAdd(&predl[row * 128 + o], wsum * val);
      }
    }
    // reduce squared error across the 16 lanes of each half (128 cols total)
#pragma unroll
    for (int r = 0; r < 8; ++r) {
      float s = sq[r];
      s += __shfl_xor(s, 1, 32);
      s += __shfl_xor(s, 2, 32);
      s += __shfl_xor(s, 4, 32);
      s += __shfl_xor(s, 8, 32);
      if (n == 0) {
        int row = t * 16 + r + 8 * half;
        float reward = -(s * (1.0f / 128.0f));
        float cnt = (seli[row * 2 + 0] == w ? 1.0f : 0.0f) +
                    (seli[row * 2 + 1] == w ? 1.0f : 0.0f);
        if (cnt != 0.0f) atomicAdd(&rsuml[w], reward * cnt);
      }
    }
  }
  __syncthreads();
  for (int i = tid; i < 32 * 128; i += 256)
    out[(base + (i >> 7)) * OO + (i & 127)] = predl[i];
  if (tid < 8) atomicAdd(&ws[8 + tid], rsuml[tid]);
}

// ---------------------------------------------------------------------------
// Kernel 3: bandit statistic finalize (E=8 elementwise)
// ---------------------------------------------------------------------------
__global__ void k_finalize(const float* __restrict__ rew,
                           const float* __restrict__ ema,
                           const int* __restrict__ pulls,
                           const float* __restrict__ ws,
                           float* __restrict__ out) {
  int t = threadIdx.x;
  if (t < 8) {
    float counts = ws[t], rs = ws[8 + t], prob = ws[16 + t];
    float pf = (float)pulls[t];
    float np = pf + counts;
    float ntr = rew[t] + rs;
    float navg = (np > 0.0f) ? ntr / fmaxf(np, 1.0f) : 0.0f;
    float nema = 0.99f * ema[t] + 0.01f * (prob * (1.0f / (float)BB));
    out[ST_OFF + t] = np;
    out[ST_OFF + 8 + t] = navg;
    out[ST_OFF + 16 + t] = nema;
  }
}

// ---------------------------------------------------------------------------
extern "C" void kernel_launch(void* const* d_in, const int* in_sizes, int n_in,
                              void* d_out, int out_size, void* d_ws,
                              size_t ws_size, hipStream_t stream) {
  const float* x       = (const float*)d_in[0];
  const float* targets = (const float*)d_in[1];
  const float* W1      = (const float*)d_in[2];
  const float* b1      = (const float*)d_in[3];
  const float* W2      = (const float*)d_in[4];
  const float* b2      = (const float*)d_in[5];
  const float* S1      = (const float*)d_in[6];
  const float* sb1     = (const float*)d_in[7];
  const float* S2      = (const float*)d_in[8];
  const float* sb2     = (const float*)d_in[9];
  const float* We      = (const float*)d_in[10];
  const float* be      = (const float*)d_in[11];
  const float* rew     = (const float*)d_in[12];
  const float* ema     = (const float*)d_in[13];
  const int*   pulls   = (const int*)d_in[14];
  const int*   total   = (const int*)d_in[15];

  float*  out = (float*)d_out;
  float*  ws  = (float*)d_ws;
  bf16_t* WeT = (bf16_t*)((char*)d_ws + 256);  // E*O*D bf16 = 512 KB scratch

  hipMemsetAsync(d_ws, 0, 128, stream);  // zero bandit accumulators
  k_transpose_we<<<(EE * OO * DD) / 256, 256, 0, stream>>>(We, WeT);
  k_router<<<BB / 64, 128, 0, stream>>>(x, W1, b1, W2, b2, S1, sb1, S2, sb2,
                                        rew, pulls, total, out, ws);
  k_expert<<<BB / 32, 256, 0, stream>>>(x, targets, WeT, be, out, ws);
  k_finalize<<<1, 32, 0, stream>>>(rew, ema, pulls, ws, out);
}
